// HyperAttention_74775380623855
// MI455X (gfx1250) — matrix operand order repaired and checked by reading, not compiled, not run on hardware
//
#include <hip/hip_runtime.h>

typedef __attribute__((ext_vector_type(16))) _Float16 v16h;
typedef __attribute__((ext_vector_type(8)))  _Float16 v8h;
typedef __attribute__((ext_vector_type(8)))  float    v8f;
typedef __attribute__((ext_vector_type(4)))  float    f4;

#define HEADS   8
#define SEQ     8192
#define NROWS   (HEADS * SEQ)   // 65536 rows per tensor
#define DDIM    64
#define NEG_BIG -3.0e38f
#define F32_EPS 1.1920929e-07f

// ---------------------------------------------------------------- helpers
static __device__ __forceinline__ v8f wmma16(v16h a, v16h b, v8f c) {
  // D = A(16x32 f16) * B(32x16 f16) + C(16x16 f32)
  return __builtin_amdgcn_wmma_f32_16x16x32_f16(false, a, false, b, (short)0, c,
                                                false, false);
}
static __device__ __forceinline__ v16h ld16(const _Float16* p) {
  v8h lo = *(const v8h*)p;
  v8h hh = *(const v8h*)(p + 8);
  v16h r;
#pragma unroll
  for (int i = 0; i < 8; ++i) { r[i] = lo[i]; r[i + 8] = hh[i]; }
  return r;
}

// max-reduce across the 16 lanes of a half-wave row (VALU v_permlane16_b32,
// avoids ds_bpermute); falls back to __shfl_xor if builtin unavailable.
#if __has_builtin(__builtin_amdgcn_permlane16)
static __device__ __forceinline__ float pl16(float x, unsigned lo, unsigned hi) {
  unsigned v = (unsigned)__float_as_int(x);
  unsigned t = __builtin_amdgcn_permlane16(v, v, lo, hi, false, false);
  return __int_as_float((int)t);
}
static __device__ __forceinline__ float xmax16(float x) {
  x = fmaxf(x, pl16(x, 0x67452301u, 0xEFCDAB89u));  // xor 1
  x = fmaxf(x, pl16(x, 0x54761032u, 0xDCFE98BAu));  // xor 2
  x = fmaxf(x, pl16(x, 0x32107654u, 0xBA98FEDCu));  // xor 4
  x = fmaxf(x, pl16(x, 0xFEDCBA98u, 0x76543210u));  // xor 8
  return x;
}
#else
static __device__ __forceinline__ float xmax16(float x) {
#pragma unroll
  for (int off = 8; off > 0; off >>= 1) x = fmaxf(x, __shfl_xor(x, off, 16));
  return x;
}
#endif

// ---------------------------------------------------------------- LSH hash
// hash(row) = gray( sum_j (row . proj[:,j] > 0) << j ),  gray(b)=b^(b>>1)==_PERM[b]
__global__ __launch_bounds__(256) void lsh_hash_kernel(
    const float* __restrict__ Q, const float* __restrict__ K,
    const float* __restrict__ P, int* __restrict__ Hq, int* __restrict__ Hk) {
  __shared__ float pj[DDIM * 7];
  for (int i = threadIdx.x; i < DDIM * 7; i += 256) pj[i] = P[i];
  __syncthreads();
  int r = blockIdx.x * 256 + threadIdx.x;           // 0 .. 2*NROWS-1
  const float* src = (r < NROWS) ? (Q + (size_t)r * DDIM)
                                 : (K + (size_t)(r - NROWS) * DDIM);
  float acc[7];
#pragma unroll
  for (int j = 0; j < 7; ++j) acc[j] = 0.f;
  for (int d = 0; d < DDIM; ++d) {
    float x = src[d];
#pragma unroll
    for (int j = 0; j < 7; ++j) acc[j] += x * pj[d * 7 + j];
  }
  int bin = 0;
#pragma unroll
  for (int j = 0; j < 7; ++j) bin |= (int)(acc[j] > 0.f) << j;
  int h = bin ^ (bin >> 1);
  if (r < NROWS) Hq[r] = h; else Hk[r - NROWS] = h;
}

// ------------------------------------------------- stable counting argsort
// seg 0..15 : level-1 (len 2048, q = v*4096+2048.., k = v*4096..)
// seg 16..23: level-0 (len 4096, q = h*8192+4096.., k = h*8192..)
__global__ __launch_bounds__(128) void lsh_sort_kernel(
    const int* __restrict__ Hq, const int* __restrict__ Hk,
    int* __restrict__ qidx1, int* __restrict__ kidx1,
    int* __restrict__ qidx0, int* __restrict__ kidx0) {
  __shared__ int hist[128];
  __shared__ int basep[128];
  int seg = blockIdx.x;
  int len, qbase, kbase;
  int *qout, *kout;
  if (seg < 16) {
    len = 2048; qbase = seg * 4096 + 2048; kbase = seg * 4096;
    qout = qidx1 + seg * 2048; kout = kidx1 + seg * 2048;
  } else {
    int hh = seg - 16;
    len = 4096; qbase = hh * 8192 + 4096; kbase = hh * 8192;
    qout = qidx0 + hh * 4096; kout = kidx0 + hh * 4096;
  }
  for (int pass = 0; pass < 2; ++pass) {
    const int* H = pass ? (Hk + kbase) : (Hq + qbase);
    int rowoff   = pass ? kbase : qbase;
    int* outp    = pass ? kout : qout;
    hist[threadIdx.x] = 0;
    __syncthreads();
    for (int i = threadIdx.x; i < len; i += 128) atomicAdd(&hist[H[i]], 1);
    __syncthreads();
    if (threadIdx.x == 0) {
      int s = 0;
      for (int b = 0; b < 128; ++b) { basep[b] = s; s += hist[b]; }
    }
    __syncthreads();
    {  // stable: thread b emits in index order for its bin
      int b = threadIdx.x, off = basep[b];
      for (int i = 0; i < len; ++i)
        if (H[i] == b) outp[off++] = rowoff + i;
    }
    __syncthreads();
  }
}

// ---------------------------------------------------------------- attention
// MODE 0: causal base (seg = virtual head 0..31, len 2048, direct rows,
//         writes d_out + lseMain at global rows)
// MODE 1: LSH block-diagonal (gathered q/k rows, 256-key block per 256-q block)
// MODE 2: sampled 256 keys with same-block bias mask, lse += log(len/256)
template <int MODE>
__global__ __launch_bounds__(256) void attn_kernel(
    const float* __restrict__ Qg, const float* __restrict__ Kg,
    const float* __restrict__ Vg,
    const int* __restrict__ qmap, const int* __restrict__ kmap,
    float* __restrict__ Oa, float* __restrict__ Ol, int level, float scale) {
  __shared__ _Float16 Klds[2][32][80];    // [buf][key][dim] (f16), padded
  __shared__ _Float16 Vt[2][64][40];      // [buf][dim][key] (f16), padded
  __shared__ _Float16 Plds[8][16][40];    // per-wave P tile [m][key]
  __shared__ int smapRow[256];
  __shared__ int sblk[256];

  const int tid = threadIdx.x;
  const int wv = tid >> 5, lane = tid & 31;
  const int n = lane & 15, hi = lane >> 4;
  const int qb = blockIdx.x, seg = blockIdx.y;

  int len, qrow0, krow0;
  if (MODE == 0)      { len = 2048; qrow0 = seg * 2048;        krow0 = seg * 2048; }
  else if (level == 1){ len = 2048; qrow0 = seg * 4096 + 2048; krow0 = seg * 4096; }
  else                { len = 4096; qrow0 = seg * 8192 + 4096; krow0 = seg * 8192; }

  if (MODE == 2) {  // deterministic pseudo-random sample positions
    unsigned u = (unsigned)tid * 2654435761u ^ ((unsigned)(level + 1) * 0x9E3779B9u);
    u ^= u >> 16; u *= 0x7feb352du; u ^= u >> 15; u *= 0x846ca68bu; u ^= u >> 16;
    int pos = (int)(u % (unsigned)len);
    smapRow[tid] = kmap[seg * len + pos];
    sblk[tid] = pos >> 8;
  }

  // ---- Q fragments (A layout: lane m = lane&15; halves -> K per hi group)
  int qpos = qb * 128 + wv * 16 + n;
  int qg = (MODE == 0) ? (qrow0 + qpos) : qmap[seg * len + qpos];
  const float* qp = Qg + (size_t)qg * DDIM;
  v16h qa[2];
#pragma unroll
  for (int f = 0; f < 2; ++f) {
    int c = f * 32 + hi * 16;
    f4 x0 = *(const f4*)(qp + c);
    f4 x1 = *(const f4*)(qp + c + 4);
    f4 y0 = *(const f4*)(qp + c + 8);
    f4 y1 = *(const f4*)(qp + c + 12);
#pragma unroll
    for (int j = 0; j < 4; ++j) {
      qa[f][j]      = (_Float16)x0[j];
      qa[f][4 + j]  = (_Float16)x1[j];
      qa[f][8 + j]  = (_Float16)y0[j];
      qa[f][12 + j] = (_Float16)y1[j];
    }
  }
  v16h ones;
#pragma unroll
  for (int i = 0; i < 16; ++i) ones[i] = (_Float16)1.0f;

  float ms[8], ls[8];
  v8f ot[4];
#pragma unroll
  for (int r = 0; r < 8; ++r) { ms[r] = -1e30f; ls[r] = 0.f; }
#pragma unroll
  for (int t = 0; t < 4; ++t)
#pragma unroll
    for (int r = 0; r < 8; ++r) ot[t][r] = 0.f;

  __syncthreads();  // smapRow/sblk visible before first staging load

  const int nchunks = (MODE == 0) ? (qb * 4 + 4) : 8;
  const int key = tid >> 3;
  const int c0t = (tid & 7) * 8;
  const int* kmB = (MODE == 1) ? (kmap + seg * len + (qb >> 1) * 256) : nullptr;

  // ---- load chunk 0 into registers (double-buffered pipeline)
  f4 ck0, ck1, cv0, cv1;
  {
    int grow;
    if (MODE == 0)      grow = krow0 + key;
    else if (MODE == 1) grow = kmB[key];
    else                grow = smapRow[key];
    const float* kp = Kg + (size_t)grow * DDIM + c0t;
    ck0 = *(const f4*)kp; ck1 = *(const f4*)(kp + 4);
    const float* vp = Vg + (size_t)grow * DDIM + c0t;
    cv0 = *(const f4*)vp; cv1 = *(const f4*)(vp + 4);
  }

  for (int ch = 0; ch < nchunks; ++ch) {
    // ---- issue next chunk's global loads (overlap with this chunk's compute)
    f4 nk0, nk1, nv0, nv1;
    if (ch + 1 < nchunks) {
      int grow;
      if (MODE == 0)      grow = krow0 + (ch + 1) * 32 + key;
      else if (MODE == 1) grow = kmB[(ch + 1) * 32 + key];
      else                grow = smapRow[(ch + 1) * 32 + key];
      const float* kp = Kg + (size_t)grow * DDIM + c0t;
      nk0 = *(const f4*)kp; nk1 = *(const f4*)(kp + 4);
      const float* vp = Vg + (size_t)grow * DDIM + c0t;
      nv0 = *(const f4*)vp; nv1 = *(const f4*)(vp + 4);
    }
    // ---- convert + store current chunk into LDS buffer ch&1
    const int bb = ch & 1;
#pragma unroll
    for (int j = 0; j < 4; ++j) {
      Klds[bb][key][c0t + j]     = (_Float16)ck0[j];
      Klds[bb][key][c0t + 4 + j] = (_Float16)ck1[j];
      Vt[bb][c0t + j][key]       = (_Float16)cv0[j];
      Vt[bb][c0t + 4 + j][key]   = (_Float16)cv1[j];
    }
    __syncthreads();

    // ---- S = Q K^T  (two 16x16 C tiles over 32 keys; K-dim 64 = 2 steps)
    v16h b00 = ld16(&Klds[bb][n][hi * 16]);
    v16h b10 = ld16(&Klds[bb][n][32 + hi * 16]);
    v16h b01 = ld16(&Klds[bb][16 + n][hi * 16]);
    v16h b11 = ld16(&Klds[bb][16 + n][32 + hi * 16]);
    v8f s0, s1;
#pragma unroll
    for (int r = 0; r < 8; ++r) { s0[r] = 0.f; s1[r] = 0.f; }
    s0 = wmma16(qa[0], b00, s0);
    s0 = wmma16(qa[1], b10, s0);
    s1 = wmma16(qa[0], b01, s1);
    s1 = wmma16(qa[1], b11, s1);

#pragma unroll
    for (int r = 0; r < 8; ++r) { s0[r] *= scale; s1[r] *= scale; }

    if (MODE == 0) {
      // wave-uniform: only diagonal tiles need the causal mask (scalar branch,
      // EXEC stays all-ones around the WMMAs)
      if (ch * 32 + 31 > qb * 128 + wv * 16) {
#pragma unroll
        for (int r = 0; r < 8; ++r) {
          int qr = qb * 128 + wv * 16 + r + 8 * hi;
          int k0p = ch * 32 + n;
          if (k0p > qr)      s0[r] = NEG_BIG;
          if (k0p + 16 > qr) s1[r] = NEG_BIG;
        }
      }
    }
    if (MODE == 2) {  // same-block bias mask
      int qblk = qb >> 1;
      int m0 = (sblk[ch * 32 + n] == qblk);
      int m1 = (sblk[ch * 32 + 16 + n] == qblk);
#pragma unroll
      for (int r = 0; r < 8; ++r) {
        if (m0) s0[r] = NEG_BIG;
        if (m1) s1[r] = NEG_BIG;
      }
    }

    // ---- online softmax (row = r + 8*hi spread over 16 lanes; max via VALU)
    float alpha[8];
#pragma unroll
    for (int r = 0; r < 8; ++r) {
      float rm = xmax16(fmaxf(s0[r], s1[r]));
      float mn = fmaxf(ms[r], rm);
      alpha[r] = __expf(ms[r] - mn);
      float p0 = __expf(s0[r] - mn);
      float p1 = __expf(s1[r] - mn);
      ms[r] = mn;
      Plds[wv][r + 8 * hi][n]      = (_Float16)p0;
      Plds[wv][r + 8 * hi][16 + n] = (_Float16)p1;
    }
    __builtin_amdgcn_wave_barrier();  // keep LDS P write->read ordered

    // ---- P fragment (A layout)
    v8h plo = *(const v8h*)&Plds[wv][n][hi * 16];
    v8h phi = *(const v8h*)&Plds[wv][n][hi * 16 + 8];
    v16h pa;
#pragma unroll
    for (int i = 0; i < 8; ++i) { pa[i] = plo[i]; pa[i + 8] = phi[i]; }

    // ---- row sums via WMMA against all-ones B (replicated in C layout)
    v8f sums;
#pragma unroll
    for (int r = 0; r < 8; ++r) sums[r] = 0.f;
    sums = wmma16(pa, ones, sums);
#pragma unroll
    for (int r = 0; r < 8; ++r) ls[r] = ls[r] * alpha[r] + sums[r];

#pragma unroll
    for (int t = 0; t < 4; ++t)
#pragma unroll
      for (int r = 0; r < 8; ++r) ot[t][r] *= alpha[r];

    // ---- O += P V  (B = V^T tiles)
#pragma unroll
    for (int t = 0; t < 4; ++t) {
      v16h bv = ld16(&Vt[bb][t * 16 + n][hi * 16]);
      ot[t] = wmma16(pa, bv, ot[t]);
    }

    ck0 = nk0; ck1 = nk1; cv0 = nv0; cv1 = nv1;
  }

  // ---- epilogue
  float inv[8];
#pragma unroll
  for (int r = 0; r < 8; ++r) inv[r] = (ls[r] > 0.f) ? (1.f / ls[r]) : 0.f;
  float logcorr = (MODE == 2) ? __logf((float)len * (1.f / 256.f)) : 0.f;
  int rowb = (MODE == 0) ? (qrow0 + qb * 128 + wv * 16) : (seg * len + qb * 128 + wv * 16);
#pragma unroll
  for (int t = 0; t < 4; ++t)
#pragma unroll
    for (int r = 0; r < 8; ++r) {
      int m = r + 8 * hi;
      Oa[(size_t)(rowb + m) * DDIM + t * 16 + n] = ot[t][r] * inv[r];
    }
  if (n == 0) {
#pragma unroll
    for (int r = 0; r < 8; ++r) {
      int m = r + 8 * hi;
      Ol[rowb + m] = ms[r] + __logf(ls[r]) + logcorr;
    }
  }
}

// ---------------------------------------------- fused _add_attn merge+scatter
// merged = add_attn(block, sampled); out[g] = add_attn(out[g], merged); g = qmap[p]
__global__ __launch_bounds__(256) void merge_scatter_kernel(
    const float* __restrict__ blkA, const float* __restrict__ blkL,
    const float* __restrict__ smpA, const float* __restrict__ smpL,
    const int* __restrict__ qmap, float* __restrict__ out,
    float* __restrict__ lseMain) {
  int idx = blockIdx.x * 256 + threadIdx.x;  // 32768 rows * 16 quads
  int row = idx >> 4;
  int c4 = (idx & 15) * 4;
  float l1 = blkL[row], l2 = smpL[row];
  float c = 1.f / (1.f + __expf(l2 - l1));
  float lnc = l1 - __logf(c + F32_EPS);
  int g = qmap[row];
  float l3 = lseMain[g];
  float c2 = 1.f / (1.f + __expf(lnc - l3));
  float lo = l3 - __logf(c2 + F32_EPS);
  const float* ba = blkA + (size_t)row * DDIM + c4;
  const float* sa = smpA + (size_t)row * DDIM + c4;
  float* op = out + (size_t)g * DDIM + c4;
#pragma unroll
  for (int j = 0; j < 4; ++j) {
    float mv = c * ba[j] + (1.f - c) * sa[j];
    op[j] = c2 * op[j] + (1.f - c2) * mv;
  }
  if (c4 == 0) lseMain[g] = lo;
}

// ------------------------------------------------------------------ driver
extern "C" void kernel_launch(void* const* d_in, const int* in_sizes, int n_in,
                              void* d_out, int out_size, void* d_ws, size_t ws_size,
                              hipStream_t stream) {
  const float* Q = (const float*)d_in[0];
  const float* K = (const float*)d_in[1];
  const float* V = (const float*)d_in[2];
  const float* P = (const float*)d_in[3];
  float* out = (float*)d_out;

  char* w = (char*)d_ws;
  float* lseMain = (float*)w; w += (size_t)NROWS * 4;
  int* Hq    = (int*)w; w += (size_t)NROWS * 4;
  int* Hk    = (int*)w; w += (size_t)NROWS * 4;
  int* qidx1 = (int*)w; w += (size_t)32768 * 4;
  int* kidx1 = (int*)w; w += (size_t)32768 * 4;
  int* qidx0 = (int*)w; w += (size_t)32768 * 4;
  int* kidx0 = (int*)w; w += (size_t)32768 * 4;
  float* blkA = (float*)w; w += (size_t)32768 * 64 * 4;
  float* blkL = (float*)w; w += (size_t)32768 * 4;
  float* smpA = (float*)w; w += (size_t)32768 * 64 * 4;
  float* smpL = (float*)w; w += (size_t)32768 * 4;

  const float scale = 0.125f;  // 64^-0.5

  lsh_hash_kernel<<<512, 256, 0, stream>>>(Q, K, P, Hq, Hk);
  lsh_sort_kernel<<<24, 128, 0, stream>>>(Hq, Hk, qidx1, kidx1, qidx0, kidx0);

  // level-2 base: causal exact attention, 32 vheads x 16 q-blocks of 128
  attn_kernel<0><<<dim3(16, 32), 256, 0, stream>>>(Q, K, V, nullptr, nullptr,
                                                   out, lseMain, 0, scale);

  for (int level = 1; level >= 0; --level) {
    dim3 grid(level ? 16 : 32, level ? 16 : 8);  // (qblocks, segments)
    const int* qm = level ? qidx1 : qidx0;
    const int* km = level ? kidx1 : kidx0;
    attn_kernel<1><<<grid, 256, 0, stream>>>(Q, K, V, qm, km, blkA, blkL, level, scale);
    attn_kernel<2><<<grid, 256, 0, stream>>>(Q, K, V, qm, km, smpA, smpL, level, scale);
    merge_scatter_kernel<<<2048, 256, 0, stream>>>(blkA, blkL, smpA, smpL, qm,
                                                   out, lseMain);
  }
  (void)in_sizes; (void)n_in; (void)out_size; (void)ws_size;
}
